// CausalSelfAttention_58205396795860
// MI455X (gfx1250) — matrix operand order repaired and checked by reading, not compiled, not run on hardware
//
#include <hip/hip_runtime.h>

#define B_  2
#define T_  2048
#define DM_ 2048
#define NH_ 16
#define HD_ 128
#define BT_ (B_ * T_)

typedef __attribute__((ext_vector_type(16))) __bf16 v16bf;
typedef __attribute__((ext_vector_type(8)))  float  v8f;

union Frag16 { v16bf v; unsigned int u[8]; };

__device__ __forceinline__ v8f v8f_zero() {
  v8f z = {0.f, 0.f, 0.f, 0.f, 0.f, 0.f, 0.f, 0.f};
  return z;
}

// round-to-nearest-even f32 -> bf16 (raw bits)
__device__ __forceinline__ unsigned short f2bf(float f) {
  unsigned int u = __float_as_uint(f);
  u += 0x7fffu + ((u >> 16) & 1u);
  return (unsigned short)(u >> 16);
}

// CDNA5 async copy: global -> LDS, 16 bytes per lane, tracked by ASYNCcnt.
// LDS offset = low 32 bits of the flat shared-memory address.
__device__ __forceinline__ void async_load_b128(void* lds_ptr, const void* gptr) {
  unsigned lds_off = (unsigned)(size_t)lds_ptr;
  unsigned long long ga = (unsigned long long)(size_t)gptr;
  asm volatile("global_load_async_to_lds_b128 %0, %1, off"
               :: "v"(lds_off), "v"(ga) : "memory");
}

__device__ __forceinline__ void wait_async0() {
  asm volatile("s_wait_asynccnt 0" ::: "memory");
}

// ---------------------------------------------------------------------------
// Conversion kernels
// ---------------------------------------------------------------------------
__global__ void f32_to_bf16_kernel(const float* __restrict__ in,
                                   unsigned short* __restrict__ out, int n) {
  for (int i = blockIdx.x * blockDim.x + threadIdx.x; i < n;
       i += gridDim.x * blockDim.x)
    out[i] = f2bf(in[i]);
}

// w stored (N out, K in) row-major fp32.  Produce pair-packed B operand:
// word[(k/2)*N + n] = { bf16(w[n][k]) , bf16(w[n][k+1]) }  (k even)
__global__ void pack_weight_kernel(const float* __restrict__ w,
                                   unsigned int* __restrict__ pk, int N, int K) {
  int total = (K >> 1) * N;
  for (int i = blockIdx.x * blockDim.x + threadIdx.x; i < total;
       i += gridDim.x * blockDim.x) {
    int k2 = i / N;
    int n  = i - k2 * N;
    unsigned int lo = f2bf(w[(size_t)n * K + 2 * k2]);
    unsigned int hi = f2bf(w[(size_t)n * K + 2 * k2 + 1]);
    pk[i] = lo | (hi << 16);
  }
}

// ---------------------------------------------------------------------------
// Generic bf16 WMMA GEMM:  C[M,N] = A[M,K] * W^T   (W given pair-packed)
// Block tile 128(M) x 64(N), 8 waves arranged 4x2, each wave 32x32 = 2x2 WMMA.
// Tile staging uses GLOBAL_LOAD_ASYNC_TO_LDS_B128 (ASYNCcnt path).
// mode 0: RoPE -> Q   bf16 [b][h][t][d]              (row-major)
// mode 1: RoPE -> K^T bf16 [b][h][(d/2)][t][2]       (pair-packed along d)
// mode 2:        -> V bf16 [b][h][(t/2)][d][2]       (pair-packed along t)
// mode 3: fp32 out [M][N]
// ---------------------------------------------------------------------------
__global__ __launch_bounds__(256)
void gemm_bf16_kernel(const unsigned int* __restrict__ A32,  // bf16 MxK as words
                      const unsigned int* __restrict__ Bpk,  // (K/2) x N words
                      int M, int N, int K, int mode,
                      const float* __restrict__ freqs,       // (T,64,2)
                      unsigned short* __restrict__ bdst,
                      float* __restrict__ fdst) {
  __shared__ unsigned int ldsA[128][16];  // [row][k-pair]  8 KB
  __shared__ unsigned int ldsB[16][64];   // [k-pair][col]  4 KB

  const int tid  = threadIdx.x;
  const int lane = tid & 31, h2 = lane >> 4, lr = lane & 15;
  const int wid  = tid >> 5;
  const int wm   = (wid & 3) * 32, wn = (wid >> 2) * 32;
  const int rowBase = blockIdx.y * 128, colBase = blockIdx.x * 64;
  const int Kw = K >> 1;

  // per-thread async-copy coordinates (constant across k-loop)
  const int ar0 = tid >> 2,          aw0 = (tid & 3) << 2;          // A chunk 0
  const int ar1 = (tid + 256) >> 2,  aw1 = aw0;                     // A chunk 1
  const int br  = tid >> 4,          bw  = (tid & 15) << 2;         // B chunk
  const unsigned int* aSrc0 = A32 + (size_t)(rowBase + ar0) * Kw + aw0;
  const unsigned int* aSrc1 = A32 + (size_t)(rowBase + ar1) * Kw + aw1;
  const unsigned int* bSrc  = Bpk + (size_t)br * N + colBase + bw;

  v8f acc[2][2];
  acc[0][0] = v8f_zero(); acc[0][1] = v8f_zero();
  acc[1][0] = v8f_zero(); acc[1][1] = v8f_zero();

  for (int kc = 0; kc < K; kc += 32) {
    const int kw = kc >> 1;
    // ---- async tile staging: 512B/wave per instruction, no VGPR round-trip
    async_load_b128(&ldsA[ar0][aw0], aSrc0 + kw);
    async_load_b128(&ldsA[ar1][aw1], aSrc1 + kw);
    async_load_b128(&ldsB[br][bw],   bSrc + (size_t)kw * N);
    wait_async0();
    __syncthreads();

    Frag16 a[2], b[2];
#pragma unroll
    for (int v = 0; v < 8; ++v) {
      // A layout: lane row = lr; K-pair word = (v<4?0:8) + h2*4 + (v&3)
      int wk = ((v & 4) << 1) + (h2 << 2) + (v & 3);
      a[0].u[v] = ldsA[wm + lr][wk];
      a[1].u[v] = ldsA[wm + 16 + lr][wk];
      // B layout: lane col = lr; K-pair word = (v<4?0:8) + h2*4 + (v&3)
      int bk = ((v & 4) << 1) + (h2 << 2) + (v & 3);
      b[0].u[v] = ldsB[bk][wn + lr];
      b[1].u[v] = ldsB[bk][wn + 16 + lr];
    }
#pragma unroll
    for (int i = 0; i < 2; ++i)
#pragma unroll
      for (int j = 0; j < 2; ++j)
        acc[i][j] = __builtin_amdgcn_wmma_f32_16x16x32_bf16(
            false, a[i].v, false, b[j].v, (short)0, acc[i][j], false, false);
    __syncthreads();
  }

  // epilogue: C layout -> lane holds col = lr, rows v + 8*h2
#pragma unroll
  for (int i = 0; i < 2; ++i) {
#pragma unroll
    for (int j = 0; j < 2; ++j) {
#pragma unroll
      for (int v = 0; v < 8; ++v) {
        int row = rowBase + wm + i * 16 + v + (h2 << 3);
        int col = colBase + wn + j * 16 + lr;
        float val = acc[i][j][v];
        if (mode == 3) {
          fdst[(size_t)row * N + col] = val;
        } else {
          int bb = row >> 11;      // row / T_
          int t  = row & (T_ - 1);
          int hh = col >> 7;
          int d  = col & 127;
          float outv = val;
          if (mode != 2) {
            // RoPE: adjacent d live in adjacent lanes
            float partner = __shfl_xor(val, 1, 32);
            const float* cs = freqs + ((size_t)t * (HD_ / 2) + (d >> 1)) * 2;
            float c = cs[0], s = cs[1];
            outv = ((d & 1) == 0) ? (val * c - partner * s)
                                  : (partner * s + val * c);
          }
          size_t headbase = (size_t)(bb * NH_ + hh) * T_ * HD_;
          size_t idx;
          if (mode == 0)
            idx = headbase + (size_t)t * HD_ + d;
          else if (mode == 1)
            idx = headbase + ((size_t)(d >> 1) * T_ + t) * 2 + (d & 1);
          else
            idx = headbase + ((size_t)(t >> 1) * HD_ + d) * 2 + (t & 1);
          bdst[idx] = f2bf(outv);
        }
      }
    }
  }
}

// ---------------------------------------------------------------------------
// Flash-style causal attention.  One wave per 16 query rows; 32-key tiles;
// online softmax; P transposed C-layout -> A-layout via per-wave LDS scratch.
// ---------------------------------------------------------------------------
__global__ __launch_bounds__(256)
void attn_fwd_kernel(const unsigned int* __restrict__ q32,   // [b][h][t][d] words
                     const unsigned int* __restrict__ k32,   // [b][h][d/2][t] words
                     const unsigned int* __restrict__ v32,   // [b][h][t/2][d] words
                     unsigned short* __restrict__ aout) {    // bf16 (B,T,DM)
  __shared__ unsigned short ldsP[8][16][32];  // per-wave P staging

  const int tid  = threadIdx.x;
  const int lane = tid & 31, h2 = lane >> 4, lr = lane & 15;
  const int wid  = tid >> 5;
  const int tilesPerSeq = T_ / 128;  // 16
  const int bh = blockIdx.x / tilesPerSeq;
  const int qt = blockIdx.x - bh * tilesPerSeq;
  const int qBase = qt * 128 + wid * 16;

  const size_t headW = (size_t)bh * T_ * (HD_ / 2);  // words per head
  const unsigned int* qh = q32 + headW;
  const unsigned int* kh = k32 + headW;
  const unsigned int* vh = v32 + headW;

  // Q fragments (A layout), kept in registers for the whole kernel
  Frag16 qf[4];
  {
    const unsigned int* qrow = qh + (size_t)(qBase + lr) * (HD_ / 2);
#pragma unroll
    for (int c = 0; c < 4; ++c)
#pragma unroll
      for (int v = 0; v < 8; ++v) {
        int wk = c * 16 + ((v & 4) << 1) + (h2 << 2) + (v & 3);
        qf[c].u[v] = qrow[wk];
      }
  }

  float m[8], l[8];
  v8f o[8];
#pragma unroll
  for (int v = 0; v < 8; ++v) { m[v] = -1e30f; l[v] = 0.f; }
#pragma unroll
  for (int f = 0; f < 8; ++f) o[f] = v8f_zero();

  const float scale = 0.08838834764831845f;  // 1/sqrt(128)
  const int ntiles = (qBase + 16 + 31) >> 5; // causal bound (per-wave uniform)

  for (int jt = 0; jt < ntiles; ++jt) {
    const int kb = jt * 32;

    // S = Q K^T for 16 rows x 32 keys (two 16x16 accumulators)
    v8f s0 = v8f_zero(), s1 = v8f_zero();
#pragma unroll
    for (int c = 0; c < 4; ++c) {
      Frag16 b0, b1;
#pragma unroll
      for (int v = 0; v < 8; ++v) {
        int dw = c * 16 + ((v & 4) << 1) + (h2 << 2) + (v & 3);  // (d>>1)
        const unsigned int* kr = kh + (size_t)dw * T_;
        b0.u[v] = kr[kb + lr];
        b1.u[v] = kr[kb + 16 + lr];
      }
      s0 = __builtin_amdgcn_wmma_f32_16x16x32_bf16(
          false, qf[c].v, false, b0.v, (short)0, s0, false, false);
      s1 = __builtin_amdgcn_wmma_f32_16x16x32_bf16(
          false, qf[c].v, false, b1.v, (short)0, s1, false, false);
    }

    // online softmax (half-wave reductions: rows split lanes 0-15 / 16-31)
#pragma unroll
    for (int v = 0; v < 8; ++v) {
      int row = qBase + v + (h2 << 3);
      int c0  = kb + lr, c1 = c0 + 16;
      float x0 = s0[v] * scale + ((c0 <= row) ? 0.f : -1e30f);
      float x1 = s1[v] * scale + ((c1 <= row) ? 0.f : -1e30f);
      float mx = fmaxf(x0, x1);
#pragma unroll
      for (int off = 8; off >= 1; off >>= 1)
        mx = fmaxf(mx, __shfl_xor(mx, off, 16));
      float nm    = fmaxf(m[v], mx);
      float alpha = __expf(m[v] - nm);
      float p0 = __expf(x0 - nm);
      float p1 = __expf(x1 - nm);
      float rs = p0 + p1;
#pragma unroll
      for (int off = 8; off >= 1; off >>= 1)
        rs += __shfl_xor(rs, off, 16);
      l[v] = l[v] * alpha + rs;
      m[v] = nm;
#pragma unroll
      for (int f = 0; f < 8; ++f) o[f][v] *= alpha;
      // stage P (C layout: row = v+8*h2, col = lr / lr+16)
      ldsP[wid][v + (h2 << 3)][lr]      = f2bf(p0);
      ldsP[wid][v + (h2 << 3)][lr + 16] = f2bf(p1);
    }

    // LDS ops from the same wave are in-order; fence counter + compiler
    asm volatile("s_wait_dscnt 0" ::: "memory");

    // reload P in A layout (row = lr, K-pairs per VGPR)
    Frag16 pf;
    {
      const unsigned int* prow = (const unsigned int*)&ldsP[wid][lr][0];
#pragma unroll
      for (int v = 0; v < 8; ++v) {
        int wk = ((v & 4) << 1) + (h2 << 2) + (v & 3);
        pf.u[v] = prow[wk];
      }
    }

    // O += P * V  (8 accumulators cover d = 0..127)
#pragma unroll
    for (int f = 0; f < 8; ++f) {
      Frag16 vb;
#pragma unroll
      for (int v = 0; v < 8; ++v) {
        int tw = (kb >> 1) + ((v & 4) << 1) + (h2 << 2) + (v & 3);  // (t>>1)
        vb.u[v] = vh[(size_t)tw * HD_ + f * 16 + lr];
      }
      o[f] = __builtin_amdgcn_wmma_f32_16x16x32_bf16(
          false, pf.v, false, vb.v, (short)0, o[f], false, false);
    }
  }

  // normalize and store bf16 into (B,T,DM)
  const int b = bh / NH_, h = bh % NH_;
#pragma unroll
  for (int v = 0; v < 8; ++v) {
    float inv = 1.f / l[v];
    int t = qBase + v + (h2 << 3);
    size_t rowb = ((size_t)b * T_ + t) * DM_ + h * HD_;
#pragma unroll
    for (int f = 0; f < 8; ++f)
      aout[rowb + f * 16 + lr] = f2bf(o[f][v] * inv);
  }
}

// ---------------------------------------------------------------------------
// Host orchestration
// ---------------------------------------------------------------------------
extern "C" void kernel_launch(void* const* d_in, const int* in_sizes, int n_in,
                              void* d_out, int out_size, void* d_ws, size_t ws_size,
                              hipStream_t stream) {
  const float* x     = (const float*)d_in[0];
  const float* freqs = (const float*)d_in[1];
  // d_in[2] = mask (unused; causal mask computed analytically)
  const float* wq    = (const float*)d_in[3];
  const float* wk    = (const float*)d_in[4];
  const float* wv    = (const float*)d_in[5];
  const float* wo    = (const float*)d_in[6];
  float* out = (float*)d_out;

  char* ws = (char*)d_ws;
  const size_t SZ_ACT = (size_t)BT_ * DM_ * 2;  // 16 MB per activation tensor
  const size_t SZ_W   = (size_t)DM_ * DM_ * 2;  // 8 MB per packed weight
  unsigned short* x_bf  = (unsigned short*)(ws);
  unsigned int*   wq_pk = (unsigned int*)(ws + SZ_ACT);
  unsigned int*   wk_pk = (unsigned int*)(ws + SZ_ACT + 1 * SZ_W);
  unsigned int*   wv_pk = (unsigned int*)(ws + SZ_ACT + 2 * SZ_W);
  unsigned int*   wo_pk = (unsigned int*)(ws + SZ_ACT + 3 * SZ_W);
  unsigned short* q_bf  = (unsigned short*)(ws + 1 * SZ_ACT + 4 * SZ_W);
  unsigned short* kT_bf = (unsigned short*)(ws + 2 * SZ_ACT + 4 * SZ_W);
  unsigned short* v_bf  = (unsigned short*)(ws + 3 * SZ_ACT + 4 * SZ_W);
  unsigned short* a_bf  = (unsigned short*)(ws + 4 * SZ_ACT + 4 * SZ_W);

  f32_to_bf16_kernel<<<8192, 256, 0, stream>>>(x, x_bf, BT_ * DM_);
  pack_weight_kernel<<<4096, 256, 0, stream>>>(wq, wq_pk, DM_, DM_);
  pack_weight_kernel<<<4096, 256, 0, stream>>>(wk, wk_pk, DM_, DM_);
  pack_weight_kernel<<<4096, 256, 0, stream>>>(wv, wv_pk, DM_, DM_);
  pack_weight_kernel<<<4096, 256, 0, stream>>>(wo, wo_pk, DM_, DM_);

  dim3 gg(DM_ / 64, BT_ / 128);  // (32, 32)
  gemm_bf16_kernel<<<gg, 256, 0, stream>>>((const unsigned int*)x_bf, wq_pk,
      BT_, DM_, DM_, /*mode=*/0, freqs, q_bf, nullptr);
  gemm_bf16_kernel<<<gg, 256, 0, stream>>>((const unsigned int*)x_bf, wk_pk,
      BT_, DM_, DM_, /*mode=*/1, freqs, kT_bf, nullptr);
  gemm_bf16_kernel<<<gg, 256, 0, stream>>>((const unsigned int*)x_bf, wv_pk,
      BT_, DM_, DM_, /*mode=*/2, freqs, v_bf, nullptr);

  attn_fwd_kernel<<<B_ * NH_ * (T_ / 128), 256, 0, stream>>>(
      (const unsigned int*)q_bf, (const unsigned int*)kT_bf,
      (const unsigned int*)v_bf, a_bf);

  gemm_bf16_kernel<<<gg, 256, 0, stream>>>((const unsigned int*)a_bf, wo_pk,
      BT_, DM_, DM_, /*mode=*/3, nullptr, nullptr, out);
}
